// FreeInt8Merger_56212531970127
// MI455X (gfx1250) — matrix operand order repaired and checked by reading, not compiled, not run on hardware
//
#include <hip/hip_runtime.h>
#include <hip/hip_bf16.h>
#include <stdint.h>

// Problem dims (fixed by the reference)
#define IN_DIM  2048
#define OUT_DIM 2048
#define HDIM    8192
#define NBATCH  4096

typedef __attribute__((ext_vector_type(16))) __bf16 v16bf;
typedef __attribute__((ext_vector_type(8)))  float  v8f;
typedef __attribute__((ext_vector_type(4)))  int    i32x4;

#define AS1 __attribute__((address_space(1)))
#define AS3 __attribute__((address_space(3)))

// Async global->LDS path (CDNA5 GLOBAL_LOAD_ASYNC_TO_LDS_B128, ASYNCcnt)
#if defined(__has_builtin)
#  if __has_builtin(__builtin_amdgcn_global_load_async_to_lds_b128)
#    define USE_ASYNC_LDS 1
#  endif
#endif
#ifndef USE_ASYNC_LDS
#  define USE_ASYNC_LDS 0
#endif

union FragU { v16bf v; uint4 q[2]; };

__device__ __forceinline__ __bf16 f2bf(float f) { return (__bf16)f; }

// 16B global -> LDS copy; async (no VGPR staging) when available.
// Builtin takes typed pointers: (i32x4 AS1*, i32x4 AS3*, imm offset, imm cpol).
// Two-step cast: bitcast in generic AS, then addrspacecast.
__device__ __forceinline__ void copy16_g2l(const __bf16* g, __bf16* l) {
#if USE_ASYNC_LDS
    i32x4* gg = (i32x4*)const_cast<__bf16*>(g);
    i32x4* ll = (i32x4*)l;
    __builtin_amdgcn_global_load_async_to_lds_b128(
        (AS1 i32x4*)gg, (AS3 i32x4*)ll, 0, 0);
#else
    *(uint4*)l = *(const uint4*)g;
#endif
}

__device__ __forceinline__ void wait_async0() {
#if USE_ASYNC_LDS
#  if __has_builtin(__builtin_amdgcn_s_wait_asynccnt)
    __builtin_amdgcn_s_wait_asynccnt(0);
#  else
    asm volatile("s_wait_asynccnt 0x0" ::: "memory");
#  endif
#endif
}

// c19 activation: matches jnp reference (even/odd floor parity sign)
__device__ __forceinline__ float c19f(float x, float craw, float rraw) {
    float c   = fmaxf(craw, 0.1f);
    float rho = fmaxf(rraw, 0.0f);
    float L   = 6.0f * c;
    float s   = x / c;
    float n   = floorf(s);
    float t   = s - n;
    float h   = t * (1.0f - t);
    float par = n - 2.0f * floorf(0.5f * n);       // 0 if floor even
    float sgn = (par == 0.0f) ? 1.0f : -1.0f;
    float interior = c * (sgn * h + rho * h * h);
    return (x >= L) ? (x - L) : ((x <= -L) ? (x + L) : interior);
}

// ---------------------------------------------------------------------------
// fp32 -> bf16 bulk convert (n multiple of 4)
// ---------------------------------------------------------------------------
__global__ __launch_bounds__(256)
void f32_to_bf16_kernel(const float* __restrict__ in, __bf16* __restrict__ out, size_t n) {
    size_t i = ((size_t)blockIdx.x * blockDim.x + threadIdx.x) * 4;
    if (i + 3 < n) {
        float4 v = *(const float4*)&in[i];
        __bf16 o[4] = { f2bf(v.x), f2bf(v.y), f2bf(v.z), f2bf(v.w) };
        *(unsigned long long*)&out[i] = *(unsigned long long*)o;   // 8B store
    }
}

// ---------------------------------------------------------------------------
// Weight materialization: codebook gather + int8 dequant + masked merge.
// Writes Wb [R x C] bf16 and Wtb [C x R] bf16 (both coalesced via LDS tile).
// Masks assumed 1 byte per element (numpy bool layout).
// ---------------------------------------------------------------------------
__global__ __launch_bounds__(256)
void materialize_w_kernel(const float* __restrict__ codebook,
                          const int*   __restrict__ idx,
                          const unsigned char* __restrict__ frozen,
                          const unsigned char* __restrict__ i8mask,
                          const float* __restrict__ i8vals,
                          const float* __restrict__ wfloat,
                          const float* __restrict__ alphap,
                          __bf16* __restrict__ Wb,
                          __bf16* __restrict__ Wtb,
                          int R, int C) {
    __shared__ float  cb[256];
    __shared__ __bf16 tile[64][66];                 // pad to dodge bank conflicts
    const int t = threadIdx.x;
    cb[t] = codebook[t];
    const float alpha = alphap[0];
    __syncthreads();

    const int R0 = blockIdx.y * 64, C0 = blockIdx.x * 64;
    #pragma unroll 4
    for (int i = 0; i < 16; i++) {
        int e = i * 256 + t;
        int r = e >> 6, c = e & 63;
        size_t gix = (size_t)(R0 + r) * C + (C0 + c);
        float w;
        if (frozen[gix])      w = cb[idx[gix] & 255];
        else if (i8mask[gix]) w = alpha * i8vals[gix];
        else                  w = wfloat[gix];
        __bf16 wb = f2bf(w);
        Wb[gix] = wb;
        tile[r][c] = wb;
    }
    __syncthreads();
    #pragma unroll 4
    for (int i = 0; i < 16; i++) {
        int e = i * 256 + t;
        int c = e >> 6, r = e & 63;                 // r fastest -> coalesced Wtb write
        Wtb[(size_t)(C0 + c) * R + (R0 + r)] = tile[r][c];
    }
}

// ---------------------------------------------------------------------------
// Tiled BF16 WMMA GEMM:  C[M,Ncols] = A[M,K] * B + bias, B given as Bt[Ncols,K]
// Block tile 128x128, BK=32 stages, double-buffered LDS filled with
// GLOBAL_LOAD_ASYNC_TO_LDS_B128; 256 threads = 8 waves (4 along M x 2 along N),
// each wave computes 32x64 via 2x4 v_wmma_f32_16x16x32_bf16 per stage.
// One barrier per stage; next-stage async loads overlap this stage's WMMAs.
// EPI: 0 = fp32 out; 1 = c19 -> bf16 out; 2 = fp32 + bf16 out; 3 = bf16 out.
// ---------------------------------------------------------------------------
template <int EPI>
__global__ __launch_bounds__(256)
void gemm_wmma_bf16(const __bf16* __restrict__ A,
                    const __bf16* __restrict__ Bt,
                    float*  __restrict__ outF,
                    __bf16* __restrict__ outB,
                    const float* __restrict__ bias,
                    const float* __restrict__ craw,
                    const float* __restrict__ rraw,
                    int M, int K, int Ncols) {
    __shared__ __align__(16) __bf16 At [2][128 * 32];
    __shared__ __align__(16) __bf16 Btl[2][128 * 32];

    const int t     = threadIdx.x;
    const int lane  = t & 31;
    const int wave  = t >> 5;
    const int g     = lane >> 4;      // lane group (K-half select)
    const int lr    = lane & 15;      // row (A) / col (B,C) within tile
    const int waveM = wave & 3;       // 4 waves * 32 rows  = 128
    const int waveN = wave >> 2;      // 2 waves * 64 cols  = 128
    const int blockN = blockIdx.x * 128;
    const int blockM = blockIdx.y * 128;

    const v8f vzero = {0.f, 0.f, 0.f, 0.f, 0.f, 0.f, 0.f, 0.f};
    v8f acc[2][4];
    #pragma unroll
    for (int mi = 0; mi < 2; mi++)
        #pragma unroll
        for (int ni = 0; ni < 4; ni++) acc[mi][ni] = vzero;

    // Issue one 128x32 stage (A and B tiles): 2 x 16B per thread per operand.
    auto issue_stage = [&](int kt, int buf) {
        const int k0 = kt * 32;
        #pragma unroll
        for (int i = 0; i < 2; i++) {
            int e = (i * 256 + t) * 8;            // bf16 element offset in tile
            int r = e >> 5, c = e & 31;
            copy16_g2l(&A [(size_t)(blockM + r) * K + k0 + c], &At [buf][e]);
            copy16_g2l(&Bt[(size_t)(blockN + r) * K + k0 + c], &Btl[buf][e]);
        }
    };

    const int KT = K >> 5;
    issue_stage(0, 0);

    for (int kt = 0; kt < KT; ++kt) {
        const int buf = kt & 1;
        wait_async0();        // this wave's stage-kt loads landed in LDS
        __syncthreads();      // publish stage kt; retire readers of buf^1
        if (kt + 1 < KT) issue_stage(kt + 1, buf ^ 1);   // overlaps WMMAs below

        // A fragment (16x32): row = lane%16, K chunks [g*8,+8) and [g*8+16,+8)
        v16bf af[2];
        #pragma unroll
        for (int mi = 0; mi < 2; mi++) {
            const __bf16* p = &At[buf][(waveM * 32 + mi * 16 + lr) * 32];
            FragU u;
            u.q[0] = *(const uint4*)(p + g * 8);
            u.q[1] = *(const uint4*)(p + g * 8 + 16);
            af[mi] = u.v;
        }
        // B fragment (32x16): col = lane%16, K chunks [g*8,+8) and [g*8+16,+8)
        v16bf bfr[4];
        #pragma unroll
        for (int ni = 0; ni < 4; ni++) {
            const __bf16* p = &Btl[buf][(waveN * 64 + ni * 16 + lr) * 32];
            FragU u;
            u.q[0] = *(const uint4*)(p + g * 8);
            u.q[1] = *(const uint4*)(p + g * 8 + 16);
            bfr[ni] = u.v;
        }
        #pragma unroll
        for (int mi = 0; mi < 2; mi++)
            #pragma unroll
            for (int ni = 0; ni < 4; ni++)
                acc[mi][ni] = __builtin_amdgcn_wmma_f32_16x16x32_bf16(
                    false, af[mi], false, bfr[ni], (short)0, acc[mi][ni], false, false);
    }

    // Epilogue. C layout: VGPR j -> row = j + 8*g, col = lane%16.
    #pragma unroll
    for (int ni = 0; ni < 4; ni++) {
        int col = blockN + waveN * 64 + ni * 16 + lr;
        float b = bias[col];
        float cc = 0.f, rr = 0.f;
        if (EPI == 1) { cc = craw[col]; rr = rraw[col]; }
        #pragma unroll
        for (int mi = 0; mi < 2; mi++) {
            int row0 = blockM + waveM * 32 + mi * 16 + g * 8;
            #pragma unroll
            for (int j = 0; j < 8; j++) {
                float v = acc[mi][ni][j] + b;
                if (EPI == 1) v = c19f(v, cc, rr);
                size_t o = (size_t)(row0 + j) * Ncols + col;
                if (EPI == 0 || EPI == 2) outF[o] = v;
                if (EPI == 1 || EPI == 2 || EPI == 3) outB[o] = f2bf(v);
            }
        }
    }
}

// ---------------------------------------------------------------------------
// kernel_launch
// Inputs (setup_inputs order):
//  0 x  1 cb_W1  2 cb_W2  3 W1_idx  4 W2_idx  5 W1_frozen  6 W2_frozen
//  7 W1_i8mask  8 W2_i8mask  9 W1_int8  10 W2_int8  11 W1_float  12 W2_float
//  13 alpha1  14 alpha2  15 b1  16 b2  17 db1  18 db2  19 c_raw  20 rho_raw
// Output: dec [N,IN] then z [N,OUT], fp32.
// Workspace use: ~302 MB of bf16 buffers.
// ---------------------------------------------------------------------------
extern "C" void kernel_launch(void* const* d_in, const int* in_sizes, int n_in,
                              void* d_out, int out_size, void* d_ws, size_t ws_size,
                              hipStream_t stream) {
    const float* x     = (const float*)d_in[0];
    const float* cb1   = (const float*)d_in[1];
    const float* cb2   = (const float*)d_in[2];
    const int*   w1idx = (const int*)d_in[3];
    const int*   w2idx = (const int*)d_in[4];
    const unsigned char* w1fro = (const unsigned char*)d_in[5];
    const unsigned char* w2fro = (const unsigned char*)d_in[6];
    const unsigned char* w1i8m = (const unsigned char*)d_in[7];
    const unsigned char* w2i8m = (const unsigned char*)d_in[8];
    const float* w1i8 = (const float*)d_in[9];
    const float* w2i8 = (const float*)d_in[10];
    const float* w1f  = (const float*)d_in[11];
    const float* w2f  = (const float*)d_in[12];
    const float* a1   = (const float*)d_in[13];
    const float* a2   = (const float*)d_in[14];
    const float* b1   = (const float*)d_in[15];
    const float* b2   = (const float*)d_in[16];
    const float* db1  = (const float*)d_in[17];
    const float* db2  = (const float*)d_in[18];
    const float* craw = (const float*)d_in[19];
    const float* rraw = (const float*)d_in[20];

    float* dec = (float*)d_out;                               // [N, IN_DIM]
    float* zf  = (float*)d_out + (size_t)NBATCH * IN_DIM;     // [N, OUT_DIM]

    // Workspace carve (all bf16, every region 16B-aligned)
    __bf16* p    = (__bf16*)d_ws;
    __bf16* W1b  = p; p += (size_t)IN_DIM * HDIM;    // W1      [IN , H ]
    __bf16* W1tb = p; p += (size_t)IN_DIM * HDIM;    // W1^T    [H  , IN]
    __bf16* W2b  = p; p += (size_t)HDIM * OUT_DIM;   // W2      [H  ,OUT]
    __bf16* W2tb = p; p += (size_t)HDIM * OUT_DIM;   // W2^T    [OUT, H ]
    __bf16* Xb   = p; p += (size_t)NBATCH * IN_DIM;  // x       [N  , IN]
    __bf16* Hb   = p; p += (size_t)NBATCH * HDIM;    // h       [N  , H ]
    __bf16* Zb   = p; p += (size_t)NBATCH * OUT_DIM; // z       [N  ,OUT]
    __bf16* Ub   = p; p += (size_t)NBATCH * HDIM;    // u       [N  , H ]

    // 1) x -> bf16
    {
        size_t n = (size_t)NBATCH * IN_DIM;
        f32_to_bf16_kernel<<<(unsigned)((n / 4 + 255) / 256), 256, 0, stream>>>(x, Xb, n);
    }
    // 2) materialize effective weights (+ transposes) in bf16
    materialize_w_kernel<<<dim3(HDIM / 64, IN_DIM / 64), 256, 0, stream>>>(
        cb1, w1idx, w1fro, w1i8m, w1i8, w1f, a1, W1b, W1tb, IN_DIM, HDIM);
    materialize_w_kernel<<<dim3(OUT_DIM / 64, HDIM / 64), 256, 0, stream>>>(
        cb2, w2idx, w2fro, w2i8m, w2i8, w2f, a2, W2b, W2tb, HDIM, OUT_DIM);

    // 3) h = c19(x @ W1 + b1)            B as [H,IN]  = W1^T
    gemm_wmma_bf16<1><<<dim3(HDIM / 128, NBATCH / 128), 256, 0, stream>>>(
        Xb, W1tb, nullptr, Hb, b1, craw, rraw, NBATCH, IN_DIM, HDIM);
    // 4) z = h @ W2 + b2  (fp32 out + bf16)   B as [OUT,H] = W2^T
    gemm_wmma_bf16<2><<<dim3(OUT_DIM / 128, NBATCH / 128), 256, 0, stream>>>(
        Hb, W2tb, zf, Zb, b2, nullptr, nullptr, NBATCH, HDIM, OUT_DIM);
    // 5) u = z @ W2^T + db1              B as [H,OUT] = W2
    gemm_wmma_bf16<3><<<dim3(HDIM / 128, NBATCH / 128), 256, 0, stream>>>(
        Zb, W2b, nullptr, Ub, db1, nullptr, nullptr, NBATCH, OUT_DIM, HDIM);
    // 6) dec = u @ W1^T + db2 (fp32 out)  B as [IN,H] = W1
    gemm_wmma_bf16<0><<<dim3(IN_DIM / 128, NBATCH / 128), 256, 0, stream>>>(
        Ub, W1b, dec, nullptr, db2, nullptr, nullptr, NBATCH, HDIM, IN_DIM);
}
